// SharedWeightBiLSTM_18631568130377
// MI455X (gfx1250) — matrix operand order repaired and checked by reading, not compiled, not run on hardware
//
#include <hip/hip_runtime.h>
#include <hip/hip_bf16.h>
#include <math.h>

typedef __bf16 bf16_t;
typedef __attribute__((ext_vector_type(16))) __bf16 v16bf;
typedef __attribute__((ext_vector_type(8)))  __bf16 v8bf;
typedef __attribute__((ext_vector_type(8)))  float  v8f;

#define SEQ   1024
#define BATCH 32
#define HID   512
#define G4    2048            // 4*HID
#define NSLICE 8              // workgroups per direction in recurrence
#define HSL    64             // h columns owned per slice (HID / NSLICE)

// ---------------------------------------------------------------------------
// WMMA helpers (CDNA5 16x16x32 bf16, f32 accumulate)
// ---------------------------------------------------------------------------
__device__ __forceinline__ v8f wmma_bf16(v16bf a, v16bf b, v8f c) {
  // 8 args: (neg_a, A, neg_b, B, c_mod, C, reuse_a, reuse_b)
  return __builtin_amdgcn_wmma_f32_16x16x32_bf16(false, a, false, b, (short)0, c,
                                                 false, false);
}

// A fragment: 16x32 (MxK) bf16 from row-major [*, lda].
// Lane L holds row M=L&15; elems 0..15 -> K = (L>>4)*16 + 0..15
// (matches B fragment).  Two contiguous 16B loads.
__device__ __forceinline__ v16bf load_a_frag(const bf16_t* __restrict__ base,
                                             int lda, int m0, int k0, int lane) {
  int m  = m0 + (lane & 15);
  int hi = lane >> 4;
  const bf16_t* p = base + (size_t)m * lda + k0 + hi * 16;
  v8bf lo  = *(const v8bf*)(p);
  v8bf hi8 = *(const v8bf*)(p + 8);
  v16bf r;
#pragma unroll
  for (int i = 0; i < 8; ++i) { r[i] = lo[i]; r[i + 8] = hi8[i]; }
  return r;
}

// B fragment: 32x16 (KxN) bf16.  Memory holds B^T row-major [N, ldk]
// (our weights are stored [gate_out, K] so row n == column n of B).
// Lane L holds column N=L&15; elems 0..15 -> K = (L>>4)*16 + v.  One 32B load.
__device__ __forceinline__ v16bf load_b_frag(const bf16_t* __restrict__ bt,
                                             int ldk, int n0, int k0, int lane) {
  int n  = n0 + (lane & 15);
  int hi = lane >> 4;
  const bf16_t* p = bt + (size_t)n * ldk + k0 + hi * 16;
  return *(const v16bf*)p;
}

// ---------------------------------------------------------------------------
// Utility kernels
// ---------------------------------------------------------------------------
__global__ void k_zero_u32(unsigned* __restrict__ p, int n) {
  int i = blockIdx.x * blockDim.x + threadIdx.x;
  if (i < n) p[i] = 0u;
}

__global__ void k_f32_to_bf16(const float* __restrict__ s, bf16_t* __restrict__ d,
                              size_t n) {
  size_t i = (size_t)blockIdx.x * blockDim.x + threadIdx.x;
  size_t stride = (size_t)gridDim.x * blockDim.x;
  for (; i < n; i += stride) d[i] = (bf16_t)s[i];
}

// out = a + b (f32); optionally also emit bf16 copy (next layer GEMM operand)
__global__ void k_sum_seq(const float* __restrict__ a, const float* __restrict__ b,
                          float* __restrict__ of, bf16_t* __restrict__ ob, size_t n) {
  size_t i = (size_t)blockIdx.x * blockDim.x + threadIdx.x;
  size_t stride = (size_t)gridDim.x * blockDim.x;
  for (; i < n; i += stride) {
    float s = a[i] + b[i];
    if (of) of[i] = s;
    if (ob) ob[i] = (bf16_t)s;
  }
}

// ---------------------------------------------------------------------------
// Input projection: C[M,N] = A[M,K] * Bt[N,K]^T   (bf16 in, f32 out)
// 8 waves/block covering a 128x128 block tile; each wave register-blocks a
// 32x64 sub-tile (2M x 4N fragments -> 8 WMMAs per 6 fragment loads, so
// operand traffic is 0.047 B/FLOP instead of 0.125 with 1 tile/wave).
// M=S*B=32768, N=2048, K=512 -> grid = 256*16 = 4096 blocks, no remainder.
// ---------------------------------------------------------------------------
__global__ __launch_bounds__(256)
void k_gemm_xw(const bf16_t* __restrict__ A, const bf16_t* __restrict__ Bt,
               float* __restrict__ C, int M, int N, int K) {
  int lane = threadIdx.x & 31;
  int wave = threadIdx.x >> 5;            // 0..7, arranged 4 (M) x 2 (N)
  int nb   = N >> 7;                      // block cols (N/128)
  int m0 = (blockIdx.x / nb) * 128 + (wave >> 1) * 32;
  int n0 = (blockIdx.x % nb) * 128 + (wave & 1) * 64;

  v8f acc[2][4] = {};
  for (int k0 = 0; k0 < K; k0 += 32) {
    v16bf a0 = load_a_frag(A, K, m0,      k0, lane);
    v16bf a1 = load_a_frag(A, K, m0 + 16, k0, lane);
    if (k0 + 32 < K)                      // keep next B chunk warm
      __builtin_prefetch(Bt + (size_t)(n0 + (lane & 15)) * K + k0 + 32, 0, 1);
#pragma unroll
    for (int j = 0; j < 4; ++j) {
      v16bf bf = load_b_frag(Bt, K, n0 + j * 16, k0, lane);
      acc[0][j] = wmma_bf16(a0, bf, acc[0][j]);
      acc[1][j] = wmma_bf16(a1, bf, acc[1][j]);
    }
  }
  // D layout: elem r -> row +r+8*(lane>>4), col +(lane&15)
  int nn = lane & 15;
  int mb = (lane >> 4) << 3;
#pragma unroll
  for (int i = 0; i < 2; ++i)
#pragma unroll
    for (int j = 0; j < 4; ++j)
#pragma unroll
      for (int r = 0; r < 8; ++r)
        C[(size_t)(m0 + i * 16 + mb + r) * N + (n0 + j * 16 + nn)] = acc[i][j][r];
}

// ---------------------------------------------------------------------------
// Recurrence.  grid = (NSLICE, 2 directions), block = 512 (16 waves).
// Slice w owns h columns [w*64, w*64+64) and gate columns g*512 + same range.
// Per step:  gates = Gx[t] + bias + h(t-1) @ Wh^T  (WMMA), then LSTM update.
// Cross-WG per-step sync via one-shot L2 atomic counters (release/acquire).
// Wh (2MB bf16) is fully L2-resident, so streaming it every step is cheap;
// the critical path is the 1024 barrier round-trips, minimized by keeping the
// per-step work to 32 WMMAs/wave and cell state c in registers.
// ---------------------------------------------------------------------------
__global__ __launch_bounds__(512)
void k_lstm_recur(const bf16_t* __restrict__ Whbf,   // [2048,512] bf16 (B^T)
                  const float*  __restrict__ Gx,     // [S*B, 2048] f32
                  const float*  __restrict__ bias,   // [2048] f32
                  bf16_t* __restrict__ hseq_bf,      // [2][S*B*H] bf16
                  float*  __restrict__ hseq_f,       // [2][S*B*H] f32
                  float*  __restrict__ out,          // d_out base (f32)
                  unsigned* __restrict__ bar,        // [2][S] counters (zeroed)
                  int layer) {
  const int dir  = blockIdx.y;
  const int hc0  = blockIdx.x * HSL;
  const int lane = threadIdx.x & 31;
  const int wave = threadIdx.x >> 5;       // 0..15
  const int gb   = wave >> 2;              // gate block 0..3 (i,f,o,g)
  const int nl0  = (wave & 3) * 16;        // col-in-slice base for this wave
  const int n0   = gb * HID + hc0 + nl0;   // global gate column tile base

  bf16_t*   hb    = hseq_bf + (size_t)dir * SEQ * BATCH * HID;
  float*    hfseq = hseq_f  + (size_t)dir * SEQ * BATCH * HID;
  unsigned* mybar = bar + dir * SEQ;

  __shared__ float preact[4][BATCH][HSL + 2];   // +2 pad: dodge bank conflicts

  // per-thread cell state: 4 elements of the [32 x 64] slice
  const int e0 = threadIdx.x * 4;          // 512 threads * 4 = 2048 = 32*64
  const int bb = e0 >> 6;                  // batch row (constant across q)
  const int j0 = e0 & 63;                  // col-in-slice
  float creg[4] = {0.f, 0.f, 0.f, 0.f};

  for (int s = 0; s < SEQ; ++s) {
    const int t = (dir == 0) ? s : (SEQ - 1 - s);

    v8f acc0 = {}, acc1 = {};
    if (s > 0) {
      const int tp = (dir == 0) ? (t - 1) : (t + 1);
      const bf16_t* Ah = hb + (size_t)tp * BATCH * HID;
#pragma unroll 4
      for (int k0 = 0; k0 < HID; k0 += 32) {
        v16bf bf = load_b_frag(Whbf, HID, n0, k0, lane);
        v16bf a0 = load_a_frag(Ah, HID, 0,  k0, lane);
        v16bf a1 = load_a_frag(Ah, HID, 16, k0, lane);
        acc0 = wmma_bf16(a0, bf, acc0);
        acc1 = wmma_bf16(a1, bf, acc1);
      }
    }

    __syncthreads();                       // prior iteration done reading LDS
    {
      int nl   = nl0 + (lane & 15);
      int mrow = (lane >> 4) << 3;
#pragma unroll
      for (int r = 0; r < 8; ++r) {
        preact[gb][mrow + r][nl]      = acc0[r];
        preact[gb][16 + mrow + r][nl] = acc1[r];
      }
    }
    __syncthreads();

    // elementwise LSTM update for this thread's 4 (b, j) cells
    const float* gx = Gx + ((size_t)t * BATCH + bb) * G4;
#pragma unroll
    for (int q = 0; q < 4; ++q) {
      int j = j0 + q;
      int c = hc0 + j;
      float pi = preact[0][bb][j] + gx[0 * HID + c] + bias[0 * HID + c];
      float pf = preact[1][bb][j] + gx[1 * HID + c] + bias[1 * HID + c];
      float po = preact[2][bb][j] + gx[2 * HID + c] + bias[2 * HID + c];
      float pg = preact[3][bb][j] + gx[3 * HID + c] + bias[3 * HID + c];
      float it = 1.f / (1.f + __expf(-pi));
      float ft = 1.f / (1.f + __expf(-pf));
      float ot = 1.f / (1.f + __expf(-po));
      float gt = tanhf(pg);
      creg[q] = ft * creg[q] + it * gt;
      float hh = ot * tanhf(creg[q]);
      size_t off = (size_t)t * BATCH * HID + (size_t)bb * HID + c;
      hb[off]    = (bf16_t)hh;             // WMMA operand for next step
      hfseq[off] = hh;                     // f32 for the fwd+bwd sum
    }

    // inter-workgroup barrier: all 8 slices of this direction finished step s
    if (s < SEQ - 1) {
      __threadfence();                     // release h writes to agent scope
      __syncthreads();
      if (threadIdx.x == 0) {
        __hip_atomic_fetch_add(&mybar[s], 1u, __ATOMIC_ACQ_REL,
                               __HIP_MEMORY_SCOPE_AGENT);
        while (__hip_atomic_load(&mybar[s], __ATOMIC_ACQUIRE,
                                 __HIP_MEMORY_SCOPE_AGENT) < (unsigned)NSLICE)
          __builtin_amdgcn_s_sleep(1);
      }
      __syncthreads();
    }
  }

  // final carries -> d_out states region
  const size_t LI = (size_t)SEQ * BATCH * HID;           // 16,777,216
  const size_t ST = (size_t)2 * BATCH * HID;             // per-state block
  float* h_f = out + LI;
  float* c_f = h_f + ST;
  float* h_b = c_f + ST;
  float* c_b = h_b + ST;
  float* hdst = ((dir == 0) ? h_f : h_b) + (size_t)layer * BATCH * HID;
  float* cdst = ((dir == 0) ? c_f : c_b) + (size_t)layer * BATCH * HID;
  const int tlast = (dir == 0) ? (SEQ - 1) : 0;
#pragma unroll
  for (int q = 0; q < 4; ++q) {
    int c = hc0 + j0 + q;
    hdst[(size_t)bb * HID + c] =
        hfseq[(size_t)tlast * BATCH * HID + (size_t)bb * HID + c];
    cdst[(size_t)bb * HID + c] = creg[q];
  }
}

// ---------------------------------------------------------------------------
// Host orchestration
// ---------------------------------------------------------------------------
extern "C" void kernel_launch(void* const* d_in, const int* in_sizes, int n_in,
                              void* d_out, int out_size, void* d_ws, size_t ws_size,
                              hipStream_t stream) {
  const float* x  = (const float*)d_in[0];   // [1024, 32, 512]
  const float* Wx = (const float*)d_in[1];   // [2, 2048, 512]
  const float* Wh = (const float*)d_in[2];   // [2, 2048, 512]
  const float* b  = (const float*)d_in[3];   // [2, 2048]
  float* out = (float*)d_out;

  const size_t nX  = (size_t)SEQ * BATCH * HID;   // 16,777,216
  const size_t nW  = (size_t)2 * G4 * HID;        // 2,097,152 per weight tensor
  const int    nB  = 2 * 2 * SEQ;                 // barrier counters

  // workspace layout (256B aligned)
  char* ws = (char*)d_ws;
  size_t off = 0;
  auto carve = [&](size_t bytes) {
    void* p = ws + off;
    off += (bytes + 255) & ~(size_t)255;
    return p;
  };
  unsigned* bar    = (unsigned*)carve((size_t)nB * 4);
  bf16_t*   Wxbf   = (bf16_t*)  carve(nW * sizeof(bf16_t));
  bf16_t*   Whbf   = (bf16_t*)  carve(nW * sizeof(bf16_t));
  bf16_t*   Xbf    = (bf16_t*)  carve(nX * sizeof(bf16_t));      // layer input
  float*    Gx     = (float*)   carve((size_t)SEQ * BATCH * G4 * 4);
  bf16_t*   hseqbf = (bf16_t*)  carve((size_t)2 * nX * sizeof(bf16_t));
  float*    hseqf  = (float*)   carve((size_t)2 * nX * 4);
  (void)ws_size; (void)in_sizes; (void)n_in; (void)out_size;

  // 0) zero barrier counters (one-shot counters, re-zeroed every call)
  k_zero_u32<<<dim3((nB + 255) / 256), dim3(256), 0, stream>>>(bar, nB);

  // 1) f32 -> bf16 conversions for WMMA operands
  k_f32_to_bf16<<<dim3(2048), dim3(256), 0, stream>>>(Wx, Wxbf, nW);
  k_f32_to_bf16<<<dim3(2048), dim3(256), 0, stream>>>(Wh, Whbf, nW);
  k_f32_to_bf16<<<dim3(4096), dim3(256), 0, stream>>>(x,  Xbf,  nX);

  const int M = SEQ * BATCH, N = G4, K = HID;
  const int gemmBlocks = (M / 128) * (N / 128);    // 128x128 tile per block

  for (int layer = 0; layer < 2; ++layer) {
    const bf16_t* WxL = Wxbf + (size_t)layer * G4 * HID;
    const bf16_t* WhL = Whbf + (size_t)layer * G4 * HID;
    const float*  bL  = b + (size_t)layer * G4;

    // 2) Gx = X @ Wx^T  (off the critical path, pure WMMA GEMM)
    k_gemm_xw<<<dim3(gemmBlocks), dim3(256), 0, stream>>>(Xbf, WxL, Gx, M, N, K);

    // 3) bidirectional recurrence (8 slices x 2 directions, persistent)
    k_lstm_recur<<<dim3(NSLICE, 2), dim3(512), 0, stream>>>(
        WhL, Gx, bL, hseqbf, hseqf, out, bar + layer * 2 * SEQ, layer);

    // 4) layer output = fwd + bwd
    if (layer == 0) {
      // feed next layer (bf16 for GEMM); f32 copy not needed
      k_sum_seq<<<dim3(4096), dim3(256), 0, stream>>>(
          hseqf, hseqf + nX, (float*)nullptr, Xbf, nX);
    } else {
      // final layer_input -> d_out (f32)
      k_sum_seq<<<dim3(4096), dim3(256), 0, stream>>>(
          hseqf, hseqf + nX, out, (bf16_t*)nullptr, nX);
    }
  }
}
